// GCNLayer_23089744183511
// MI455X (gfx1250) — compile-verified, matrix operand order checked by reading, not run on hardware
//
#include <hip/hip_runtime.h>
#include <hip/hip_bf16.h>

#define IN_DIM 512
#define OUT_DIM 256
#define BN_EPS 1e-5f

typedef __attribute__((ext_vector_type(16))) __bf16 v16bf;
typedef __attribute__((ext_vector_type(8)))  __bf16 v8bf;
typedef __attribute__((ext_vector_type(8)))  float  v8f;
typedef __attribute__((ext_vector_type(8)))  float  v8ff;
typedef __attribute__((ext_vector_type(8)))  unsigned int v8u;

// ---------------------------------------------------------------------------
// Kernel A: W [512][256] fp32 (K-major) -> Wt [256][512] bf16 (N-major)
// ---------------------------------------------------------------------------
__global__ __launch_bounds__(256) void convert_w_kernel(
    const float* __restrict__ W, __bf16* __restrict__ Wt)
{
    int i = blockIdx.x * 256 + threadIdx.x;     // over IN_DIM*OUT_DIM
    int k = i / OUT_DIM;
    int n = i % OUT_DIM;
    Wt[(size_t)n * IN_DIM + k] = (__bf16)W[i];  // fptrunc, RTE
}

// ---------------------------------------------------------------------------
// Kernel B: support = x @ W + b via v_wmma_f32_16x16x32_bf16
// One wave computes a 16(M) x 128(N) tile: A fragment reused across 8 N-tiles.
// A-fragment per-lane layout (16-bit A 16x32, ISA 7.12.2):
//   lane<16 : M=lane,    K = [kb..kb+7] in e0..7,  [kb+16..kb+23] in e8..15, kb=k0
//   lane>=16: M=lane-16, same but kb=k0+8
// B stored as Wt[N][K] so B fragment uses identical indexing with N for M.
// ---------------------------------------------------------------------------
__global__ __launch_bounds__(256) void gemm_wmma_kernel(
    const float* __restrict__ x, const __bf16* __restrict__ Wt,
    const float* __restrict__ bias, float* __restrict__ support, int nrows)
{
    const int lane  = threadIdx.x & 31;
    const int wvblk = threadIdx.x >> 5;
    const int gwave = blockIdx.x * 8 + wvblk;
    const int mtile = gwave >> 1;          // 16-row tile
    const int ngrp  = gwave & 1;           // 128-col group
    const int row0  = mtile * 16;
    if (row0 >= nrows) return;

    const int hi = lane >> 4;              // 0/1 lane-half
    const int m  = lane & 15;

    const float* arow = x + (size_t)(row0 + m) * IN_DIM;

    v8f acc[8] = {};

    for (int k0 = 0; k0 < IN_DIM; k0 += 32) {
        const int kb = k0 + hi * 8;

        // ---- A fragment: 2x 8 contiguous fp32, hardware-converted to bf16 ----
        v16bf afrag;
        {
            v8ff f0 = *(const v8ff*)(arow + kb);        // K kb..kb+7
            v8ff f1 = *(const v8ff*)(arow + kb + 16);   // K kb+16..kb+23
            v8bf b0 = __builtin_convertvector(f0, v8bf);
            v8bf b1 = __builtin_convertvector(f1, v8bf);
            #pragma unroll
            for (int i = 0; i < 8; ++i) { afrag[i] = b0[i]; afrag[i + 8] = b1[i]; }
        }

        // ---- 8 N-tiles share this A fragment ----
        #pragma unroll
        for (int t = 0; t < 8; ++t) {
            const int ncol = ngrp * 128 + t * 16 + m;
            const __bf16* brow = Wt + (size_t)ncol * IN_DIM + kb; // 16B aligned
            uint4 lo  = *(const uint4*)(brow);        // K kb..kb+7
            uint4 hi4 = *(const uint4*)(brow + 16);   // K kb+16..kb+23
            v8u pk;
            pk[0] = lo.x;  pk[1] = lo.y;  pk[2] = lo.z;  pk[3] = lo.w;
            pk[4] = hi4.x; pk[5] = hi4.y; pk[6] = hi4.z; pk[7] = hi4.w;
            v16bf bfrag = __builtin_bit_cast(v16bf, pk);

            acc[t] = __builtin_amdgcn_wmma_f32_16x16x32_bf16(
                false, afrag, false, bfrag, (short)0, acc[t], false, false);
        }
    }

    // ---- store C + bias. C layout: VGPR r -> (lane<16: M=r, N=lane) /
    //      (lane>=16: M=r+8, N=lane-16) ----
    const int rbase = row0 + hi * 8;
    #pragma unroll
    for (int t = 0; t < 8; ++t) {
        const int ncol = ngrp * 128 + t * 16 + m;
        const float bv = bias[ncol];
        #pragma unroll
        for (int r = 0; r < 8; ++r) {
            support[(size_t)(rbase + r) * OUT_DIM + ncol] = acc[t][r] + bv;
        }
    }
}

// ---------------------------------------------------------------------------
// Kernel C: edge scatter. One wave per edge; lane handles 8 columns
// (2x 128-bit gathers from support, 8 fp32 global atomic adds).
// support + out accumulator together ~fit the 192MB L2.
// Edge index forced to SGPR so metadata comes in as scalar loads.
// ---------------------------------------------------------------------------
__global__ __launch_bounds__(256) void spmm_scatter_kernel(
    const float* __restrict__ support, const int* __restrict__ erow,
    const int* __restrict__ ecol, const float* __restrict__ eval,
    float* __restrict__ out, int E)
{
    const int gw = __builtin_amdgcn_readfirstlane(blockIdx.x * 8 + (threadIdx.x >> 5));
    if (gw >= E) return;
    const int lane = threadIdx.x & 31;

    const int   dst = erow[gw];
    const int   src = ecol[gw];
    const float w   = eval[gw];

    const float4* sp = (const float4*)(support + (size_t)src * OUT_DIM) + lane * 2;
    const float4 a = sp[0];
    const float4 b = sp[1];

    float* op = out + (size_t)dst * OUT_DIM + lane * 8;
    atomicAdd(op + 0, a.x * w);
    atomicAdd(op + 1, a.y * w);
    atomicAdd(op + 2, a.z * w);
    atomicAdd(op + 3, a.w * w);
    atomicAdd(op + 4, b.x * w);
    atomicAdd(op + 5, b.y * w);
    atomicAdd(op + 6, b.z * w);
    atomicAdd(op + 7, b.w * w);
}

// ---------------------------------------------------------------------------
// Kernel D: BatchNorm1d (eval) + exact GELU, in place on d_out. float4/thread.
// ---------------------------------------------------------------------------
__global__ __launch_bounds__(256) void bn_gelu_kernel(
    float* __restrict__ out,
    const float* __restrict__ gamma, const float* __restrict__ beta,
    const float* __restrict__ rmean, const float* __restrict__ rvar,
    int total4)
{
    int i = blockIdx.x * 256 + threadIdx.x;
    if (i >= total4) return;
    float4 v = ((float4*)out)[i];
    const int c0 = (i * 4) & (OUT_DIM - 1);   // 4 consecutive channels

    float r[4] = {v.x, v.y, v.z, v.w};
    #pragma unroll
    for (int j = 0; j < 4; ++j) {
        const int c = c0 + j;
        float inv = rsqrtf(rvar[c] + BN_EPS);
        float y = (r[j] - rmean[c]) * inv * gamma[c] + beta[c];
        r[j] = 0.5f * y * (1.0f + erff(y * 0.70710678118654752f));
    }
    ((float4*)out)[i] = make_float4(r[0], r[1], r[2], r[3]);
}

// ---------------------------------------------------------------------------
extern "C" void kernel_launch(void* const* d_in, const int* in_sizes, int n_in,
                              void* d_out, int out_size, void* d_ws, size_t ws_size,
                              hipStream_t stream) {
    const float* x     = (const float*)d_in[0];
    const int*   erow  = (const int*)d_in[1];
    const int*   ecol  = (const int*)d_in[2];
    const float* eval  = (const float*)d_in[3];
    const float* W     = (const float*)d_in[4];
    const float* bias  = (const float*)d_in[5];
    const float* gamma = (const float*)d_in[6];
    const float* beta  = (const float*)d_in[7];
    const float* rmean = (const float*)d_in[8];
    const float* rvar  = (const float*)d_in[9];

    const int nrows = in_sizes[0] / IN_DIM;   // 100000
    const int E     = in_sizes[1];            // 3200000
    float* out = (float*)d_out;

    // workspace layout: [Wt bf16 256KB][support fp32 102.4MB]
    __bf16* Wt      = (__bf16*)d_ws;
    float*  support = (float*)((char*)d_ws + (size_t)IN_DIM * OUT_DIM * sizeof(__bf16));

    // A) W -> Wt (bf16, N-major)
    convert_w_kernel<<<(IN_DIM * OUT_DIM) / 256, 256, 0, stream>>>(W, Wt);

    // zero output accumulator (capturable async memset)
    hipMemsetAsync(out, 0, (size_t)nrows * OUT_DIM * sizeof(float), stream);

    // B) GEMM: 6250 M-tiles * 2 N-groups = 12500 waves, 8 waves/block
    const int gemm_waves  = ((nrows + 15) / 16) * 2;
    const int gemm_blocks = (gemm_waves + 7) / 8;
    gemm_wmma_kernel<<<gemm_blocks, 256, 0, stream>>>(x, Wt, bias, support, nrows);

    // C) edge scatter: one wave per edge
    const int sp_blocks = (E + 7) / 8;
    spmm_scatter_kernel<<<sp_blocks, 256, 0, stream>>>(support, erow, ecol, eval, out, E);

    // D) BN + exact GELU in place, float4 per thread
    const int total4 = (int)((size_t)nrows * OUT_DIM / 4);
    bn_gelu_kernel<<<(total4 + 255) / 256, 256, 0, stream>>>(
        out, gamma, beta, rmean, rvar, total4);
}
